// CONCATNet_7447473291802
// MI455X (gfx1250) — compile-verified, matrix operand order checked by reading, not run on hardware
//
#include <hip/hip_runtime.h>

// ---------------- types ----------------
typedef __attribute__((ext_vector_type(16))) __bf16 v16bf;
typedef __attribute__((ext_vector_type(8)))  __bf16 v8bf;
typedef __attribute__((ext_vector_type(8)))  float  v8f;

// ---------------- problem constants ----------------
constexpr int NB  = 2048;   // batch
constexpr int TT  = 32;     // T
constexpr int SS  = 6;      // S
constexpr int LL  = 12;     // L
constexpr int DD  = 256;    // D
constexpr int K3D = 768;    // 3*D
constexpr int NACT = 42;    // T + (L-2)
constexpr int NKSTEP = K3D / 32;

// ---------------- workspace layout (bytes) ----------------
constexpr size_t O_DUR  = 0;                                   // f32 [B][11]
constexpr size_t O_PME  = O_DUR  + (size_t)NB * 11 * 4;        // bf16 [B][10][256]
constexpr size_t O_CTX  = O_PME  + (size_t)NB * 10 * DD * 2;   // bf16 [B][768]
constexpr size_t O_HID  = O_CTX  + (size_t)NB * K3D * 2;       // bf16 [B][256]
constexpr size_t O_GR   = O_HID  + (size_t)NB * DD * 2;        // bf16 [B][256]
constexpr size_t O_Q    = O_GR   + (size_t)NB * DD * 2;        // f32  [B][256]
constexpr size_t O_WPM  = O_Q    + (size_t)NB * DD * 4;        // bf16 [256][768]  (N,K)
constexpr size_t O_WKVS = O_WPM  + (size_t)DD * K3D * 2;       // bf16 [768][768]
constexpr size_t O_WC1  = O_WKVS + (size_t)K3D * K3D * 2;      // bf16 [256][768]
constexpr size_t O_WC2  = O_WC1  + (size_t)DD * K3D * 2;       // bf16 [256][256]
constexpr size_t O_WQ1  = O_WC2  + (size_t)DD * DD * 2;        // bf16 [256][256]
constexpr size_t O_END  = O_WQ1  + (size_t)DD * DD * 2;

// ---------------- fused-kernel dynamic LDS layout ----------------
// GEMM phase (double buffered):
constexpr int OFS_A0 = 0;          // [48][40] bf16 = 3840 B per buffer
constexpr int A_BUFSZ = 3840;
constexpr int OFS_B0 = 7680;       // [768][40] bf16 = 61440 B per buffer
constexpr int B_BUFSZ = 61440;     // staging ends at 130560
// Epilogue (aliases dead staging):
constexpr int OFS_KVS = 0;         // f32 [42][768] = 129024 B
constexpr int OFS_Q   = 129024;    // f32 [256]
constexpr int OFS_S   = 130048;    // f32 [16][42]
constexpr int OFS_MH  = 132864;    // f32 [256]
constexpr int OFS_MH2 = 133888;    // f32 [256]
constexpr int OFS_RED = 134912;    // f32 [2]
constexpr int FUSED_SMEM = 135168;

// ---------------- WMMA helpers ----------------
__device__ __forceinline__ v8f zero8() {
  v8f z;
#pragma unroll
  for (int i = 0; i < 8; ++i) z[i] = 0.0f;
  return z;
}

__device__ __forceinline__ v8f wmma_bf16(v16bf a, v16bf b, v8f c) {
  // D = A(16x32 bf16) x B(32x16 bf16) + C(16x16 f32)
  return __builtin_amdgcn_wmma_f32_16x16x32_bf16(false, a, false, b, (short)0, c, false, false);
}

// Load a 16x32 bf16 fragment from LDS (row-major, `stride` in halves).
// ISA layout: lanes 0-15 -> row=lane, K {0..7} and {16..23};
//             lanes 16-31 -> row=lane-16, K {8..15} and {24..31}.
__device__ __forceinline__ v16bf ld_frag(const __bf16* base, int stride) {
  int lane = threadIdx.x & 31;
  int r = lane & 15;
  int ko = (lane >> 4) << 3;
  const __bf16* p = base + r * stride + ko;
  union { v16bf v; v8bf h[2]; } u;
  u.h[0] = *(const v8bf*)(p);
  u.h[1] = *(const v8bf*)(p + 16);
  return u.v;
}

// ---------------- K0: f32 (K,N) -> bf16 transposed (N,K) ----------------
__global__ void wconv_kernel(const float* __restrict__ src, __bf16* __restrict__ dst,
                             int K, int N, int dld, int coff) {
  int i = blockIdx.x * 256 + threadIdx.x;
  if (i >= K * N) return;
  int n = i / K, k = i % K;
  dst[(size_t)(n + coff) * dld + k] = (__bf16)src[(size_t)k * N + n];
}

// ---------------- K1: per-batch scheduling scalars -> dur[B][11] ----------------
__global__ void dur_kernel(const float* __restrict__ clock,
                           const float* __restrict__ proc_end,
                           const float* __restrict__ purge_end,
                           const int* __restrict__ loc_stage,
                           const int* __restrict__ loc_status,
                           const int* __restrict__ robot_loc,
                           float* __restrict__ dur) {
  int b = blockIdx.x * 256 + threadIdx.x;
  if (b >= NB) return;
  float ck = clock[b];
  int rl = robot_loc[b];
  float eat[LL], base[LL];
  int stg[LL];
#pragma unroll
  for (int i = 0; i < LL; ++i) {
    float pkup = ck + 3.0f * (i != rl ? 1.0f : 0.0f);
    float eust = fmaxf(proc_end[b * LL + i], pkup);
    eat[i] = eust + 2.0f + 3.0f;                       // + UNLOAD + MOVE
    int st = loc_status[b * LL + i];
    base[i] = (st == 0) ? 0.0f
            : (st == 1) ? fmaxf(purge_end[b * LL + i] - eat[i], 0.0f)
                        : 1.0f;
    stg[i] = loc_stage[b * LL + i];
  }
#pragma unroll
  for (int i = 0; i < LL - 1; ++i) {
    float d = 1.0f;
#pragma unroll
    for (int j = 0; j < LL; ++j)
      if (stg[j] == stg[i] + 1) d = fminf(d, base[j]);
    dur[b * 11 + i] = eat[i] + d + 2.0f - ck;          // + LOAD, minus clock
  }
}

// ---------------- K2: pm_emb GEMM (M=B*10, K=768, N=256), A gathered ----------------
__global__ __launch_bounds__(256) void pm_emb_gemm(
    const float* __restrict__ enc_row, const float* __restrict__ enc_col,
    const float* __restrict__ clock, const float* __restrict__ proc_end,
    const float* __restrict__ purge_end, const float* __restrict__ W_pm_dyn,
    const int* __restrict__ pm_lot_idx, const int* __restrict__ loc_stage,
    const __bf16* __restrict__ Wpm_t, __bf16* __restrict__ pm_emb) {
  __shared__ __align__(16) __bf16 sA[128][40];
  __shared__ __align__(16) __bf16 sB[64][40];
  const int t = threadIdx.x, wave = t >> 5, lane = t & 31;
  const int m0 = blockIdx.x * 128;
  const int n0 = blockIdx.y * 64;
  v8f acc[4];
#pragma unroll
  for (int j = 0; j < 4; ++j) acc[j] = zero8();

  for (int kb = 0; kb < K3D; kb += 32) {
    __syncthreads();
    const int seg = kb >> 8;
    const int kk = (kb & 255) + ((t & 1) << 4);
    {
      int row = t >> 1, off = (t & 1) << 4;
      int r = m0 + row;
      int b = r / 10, l = r % 10;
      float vals[16];
      if (seg == 0) {                                   // pm_stage gather
        int stg = loc_stage[b * LL + l + 1];
        const float* src = enc_col + ((size_t)b * SS + stg) * DD + kk;
#pragma unroll
        for (int i = 0; i < 16; ++i) vals[i] = src[i];
      } else if (seg == 1) {                            // pm_wafer gather
        int idx = pm_lot_idx[b * 10 + l];
        if (idx < TT) {
          const float* src = enc_row + ((size_t)b * TT + idx) * DD + kk;
#pragma unroll
          for (int i = 0; i < 16; ++i) vals[i] = src[i];
        } else {
#pragma unroll
          for (int i = 0; i < 16; ++i) vals[i] = 0.0f;
        }
      } else {                                          // pm_dyn @ W_pm_dyn
        float ck = clock[b];
        float rp = fmaxf(proc_end[b * LL + l + 1] - ck, 0.0f);
        float ru = fmaxf(purge_end[b * LL + l + 1] - ck, 0.0f);
#pragma unroll
        for (int i = 0; i < 16; ++i)
          vals[i] = rp * W_pm_dyn[kk + i] + ru * W_pm_dyn[DD + kk + i];
      }
#pragma unroll
      for (int i = 0; i < 16; ++i) sA[row][off + i] = (__bf16)vals[i];
    }
    {
      int row = t >> 2, off = (t & 3) << 3;
      *(v8bf*)&sB[row][off] = *(const v8bf*)(Wpm_t + (size_t)(n0 + row) * K3D + kb + off);
    }
    __syncthreads();
    v16bf a = ld_frag(&sA[wave * 16][0], 40);
    v16bf bfr[4];
#pragma unroll
    for (int j = 0; j < 4; ++j) bfr[j] = ld_frag(&sB[j * 16][0], 40);
#pragma unroll
    for (int j = 0; j < 4; ++j) acc[j] = wmma_bf16(a, bfr[j], acc[j]);
  }
  int rbase = m0 + wave * 16 + ((lane >> 4) << 3);
  int cin = lane & 15;
#pragma unroll
  for (int j = 0; j < 4; ++j) {
    int col = n0 + j * 16 + cin;
#pragma unroll
    for (int v = 0; v < 8; ++v)
      pm_emb[(size_t)(rbase + v) * DD + col] = (__bf16)acc[j][v];
  }
}

// ---------------- K3: ctx = [mean_row | mean_col | mean_pm_emb] (bf16) ----------------
__global__ void ctx_build(const float* __restrict__ enc_row,
                          const float* __restrict__ enc_col,
                          const __bf16* __restrict__ pm_emb,
                          __bf16* __restrict__ ctxh) {
  int b = blockIdx.x, d = threadIdx.x;
  float s = 0.0f;
  for (int i = 0; i < TT; ++i) s += enc_row[((size_t)b * TT + i) * DD + d];
  ctxh[(size_t)b * K3D + d] = (__bf16)(s * (1.0f / TT));
  s = 0.0f;
  for (int i = 0; i < SS; ++i) s += enc_col[((size_t)b * SS + i) * DD + d];
  ctxh[(size_t)b * K3D + DD + d] = (__bf16)(s * (1.0f / SS));
  s = 0.0f;
  for (int l = 0; l < 10; ++l) s += (float)pm_emb[((size_t)b * 10 + l) * DD + d];
  ctxh[(size_t)b * K3D + 2 * DD + d] = (__bf16)(s * 0.1f);
}

// ---------------- K4: generic WMMA GEMM (M%128==0, N%64==0, K%32==0) ----------------
__global__ __launch_bounds__(256) void gemm_bf16(
    const __bf16* __restrict__ A, int lda,
    const __bf16* __restrict__ Bt, int ldb,
    const float* __restrict__ bias, int relu,
    float* __restrict__ outF, __bf16* __restrict__ outH,
    int M, int N, int K) {
  __shared__ __align__(16) __bf16 sA[128][40];
  __shared__ __align__(16) __bf16 sB[64][40];
  const int t = threadIdx.x, wave = t >> 5, lane = t & 31;
  const int m0 = blockIdx.x * 128;
  const int n0 = blockIdx.y * 64;
  v8f acc[4];
#pragma unroll
  for (int j = 0; j < 4; ++j) acc[j] = zero8();

  for (int kb = 0; kb < K; kb += 32) {
    __syncthreads();
    {
      int row = t >> 1, off = (t & 1) << 4;
      const __bf16* src = A + (size_t)(m0 + row) * lda + kb + off;
      *(v8bf*)&sA[row][off]     = *(const v8bf*)(src);
      *(v8bf*)&sA[row][off + 8] = *(const v8bf*)(src + 8);
    }
    {
      int row = t >> 2, off = (t & 3) << 3;
      *(v8bf*)&sB[row][off] = *(const v8bf*)(Bt + (size_t)(n0 + row) * ldb + kb + off);
    }
    __syncthreads();
    v16bf a = ld_frag(&sA[wave * 16][0], 40);
    v16bf bfr[4];
#pragma unroll
    for (int j = 0; j < 4; ++j) bfr[j] = ld_frag(&sB[j * 16][0], 40);
#pragma unroll
    for (int j = 0; j < 4; ++j) acc[j] = wmma_bf16(a, bfr[j], acc[j]);
  }
  int rbase = m0 + wave * 16 + ((lane >> 4) << 3);
  int cin = lane & 15;
#pragma unroll
  for (int j = 0; j < 4; ++j) {
    int col = n0 + j * 16 + cin;
    float bv = bias ? bias[col] : 0.0f;
#pragma unroll
    for (int v = 0; v < 8; ++v) {
      float x = acc[j][v] + bv;
      if (relu) x = fmaxf(x, 0.0f);
      if (outF) outF[(size_t)(rbase + v) * N + col] = x;
      else      outH[(size_t)(rbase + v) * N + col] = (__bf16)x;
    }
  }
}

// ---------------- K5 helpers ----------------
// Gather one 16-wide chunk of the implicit `action` matrix row (never materialized).
__device__ __forceinline__ void gather_action16(
    float* vals, int b, int row, int kb, int off,
    const float* __restrict__ enc_row, const __bf16* __restrict__ pm_emb,
    const float* __restrict__ dur, const float* __restrict__ W_time,
    const int* __restrict__ pm_lot_idx, float dur0) {
  const int seg = kb >> 8;
  const int kk = (kb & 255) + off;
  if (row >= NACT) {
#pragma unroll
    for (int i = 0; i < 16; ++i) vals[i] = 0.0f;
  } else if (row < TT) {                    // ll rows: [0 | enc_row | ll_dur]
    if (seg == 0) {
#pragma unroll
      for (int i = 0; i < 16; ++i) vals[i] = 0.0f;
    } else if (seg == 1) {
      const float* src = enc_row + ((size_t)b * TT + row) * DD + kk;
#pragma unroll
      for (int i = 0; i < 16; ++i) vals[i] = src[i];
    } else {
#pragma unroll
      for (int i = 0; i < 16; ++i) vals[i] = dur0 * W_time[kk + i];
    }
  } else {                                  // pm rows: [pm_emb | pm_wafer | pm_dur]
    int l = row - TT;
    if (seg == 0) {
      const __bf16* src = pm_emb + ((size_t)b * 10 + l) * DD + kk;
#pragma unroll
      for (int i = 0; i < 16; ++i) vals[i] = (float)src[i];
    } else if (seg == 1) {
      int idx = pm_lot_idx[b * 10 + l];
      if (idx < TT) {
        const float* src = enc_row + ((size_t)b * TT + idx) * DD + kk;
#pragma unroll
        for (int i = 0; i < 16; ++i) vals[i] = src[i];
      } else {
#pragma unroll
        for (int i = 0; i < 16; ++i) vals[i] = 0.0f;
      }
    } else {
      float dl = dur[b * 11 + 1 + l];
#pragma unroll
      for (int i = 0; i < 16; ++i) vals[i] = dl * W_time[kk + i];
    }
  }
}

// ---------------- K5: fused kvs GEMM + attention + logits, one WG per batch ----------------
__global__ __launch_bounds__(256) void fused_kvs_attn(
    const float* __restrict__ enc_row, const __bf16* __restrict__ pm_emb,
    const float* __restrict__ dur, const float* __restrict__ W_time,
    const int* __restrict__ pm_lot_idx, const __bf16* __restrict__ Wkvs_t,
    const float* __restrict__ qv, const float* __restrict__ W_mhc,
    const float* __restrict__ b_mhc, float* __restrict__ out) {
  extern __shared__ __align__(16) char smem[];
  float* sKVS = (float*)(smem + OFS_KVS);
  float* sQ   = (float*)(smem + OFS_Q);
  float* sS   = (float*)(smem + OFS_S);
  float* sMH  = (float*)(smem + OFS_MH);
  float* sMH2 = (float*)(smem + OFS_MH2);
  float* sRed = (float*)(smem + OFS_RED);

  const int b = blockIdx.x;
  const int t = threadIdx.x, wave = t >> 5, lane = t & 31;
  const float dur0 = dur[b * 11];
  const int arow = t >> 1, aoff = (t & 1) << 4;        // A-staging role (t<96)

  v8f acc[3][6];
#pragma unroll
  for (int m = 0; m < 3; ++m)
#pragma unroll
    for (int j = 0; j < 6; ++j) acc[m][j] = zero8();

  // ---- prologue: stage chunk 0 directly into buffer 0 ----
  if (t < 96) {
    float vals[16];
    gather_action16(vals, b, arow, 0, aoff, enc_row, pm_emb, dur, W_time, pm_lot_idx, dur0);
    __bf16* dst = (__bf16*)(smem + OFS_A0) + arow * 40 + aoff;
#pragma unroll
    for (int i = 0; i < 16; ++i) dst[i] = (__bf16)vals[i];
  }
#pragma unroll
  for (int rr = 0; rr < 3; ++rr) {
    int row = t + rr * 256;
    const __bf16* src = Wkvs_t + (size_t)row * K3D;
    __bf16* dst = (__bf16*)(smem + OFS_B0) + row * 40;
#pragma unroll
    for (int p = 0; p < 4; ++p) *(v8bf*)(dst + 8 * p) = *(const v8bf*)(src + 8 * p);
  }

  // ---- main loop: register-staged double buffering ----
  for (int kt = 0; kt < NKSTEP; ++kt) {
    const int cur = kt & 1;
    const int kbn = (kt + 1) * 32;                     // next chunk
    const bool haveNext = (kt + 1 < NKSTEP);
    __syncthreads();                                   // buffer `cur` is ready

    // issue next-chunk global loads into registers (hidden behind WMMAs below)
    union { v8bf h[2]; __bf16 e[16]; } ra;
    v8bf rb[12];
    if (haveNext) {
      if (t < 96) {
        float vals[16];
        gather_action16(vals, b, arow, kbn, aoff, enc_row, pm_emb, dur, W_time, pm_lot_idx, dur0);
#pragma unroll
        for (int i = 0; i < 16; ++i) ra.e[i] = (__bf16)vals[i];
      }
#pragma unroll
      for (int rr = 0; rr < 3; ++rr) {
        int row = t + rr * 256;
        const __bf16* src = Wkvs_t + (size_t)row * K3D + kbn;
#pragma unroll
        for (int p = 0; p < 4; ++p) rb[rr * 4 + p] = *(const v8bf*)(src + 8 * p);
        if (kbn + 32 < K3D)                             // gfx1250 global_prefetch_b8 path
          __builtin_prefetch(src + 32, 0, 1);
      }
    }

    // compute on buffer `cur`: cluster loads, then 18 back-to-back WMMAs
    {
      const __bf16* cA = (const __bf16*)(smem + OFS_A0 + cur * A_BUFSZ);
      const __bf16* cB = (const __bf16*)(smem + OFS_B0 + cur * B_BUFSZ);
      v16bf a0 = ld_frag(cA, 40);
      v16bf a1 = ld_frag(cA + 640, 40);
      v16bf a2 = ld_frag(cA + 1280, 40);
      v16bf bfr[6];
#pragma unroll
      for (int j = 0; j < 6; ++j) bfr[j] = ld_frag(cB + (size_t)(wave * 6 + j) * 640, 40);
#pragma unroll
      for (int j = 0; j < 6; ++j) {
        acc[0][j] = wmma_bf16(a0, bfr[j], acc[0][j]);
        acc[1][j] = wmma_bf16(a1, bfr[j], acc[1][j]);
        acc[2][j] = wmma_bf16(a2, bfr[j], acc[2][j]);
      }
    }

    // spill staged registers into the other buffer
    if (haveNext) {
      const int nxt = cur ^ 1;
      if (t < 96) {
        __bf16* dst = (__bf16*)(smem + OFS_A0 + nxt * A_BUFSZ) + arow * 40 + aoff;
        *(v8bf*)(dst)     = ra.h[0];
        *(v8bf*)(dst + 8) = ra.h[1];
      }
#pragma unroll
      for (int rr = 0; rr < 3; ++rr) {
        __bf16* dst = (__bf16*)(smem + OFS_B0 + nxt * B_BUFSZ) + (t + rr * 256) * 40;
#pragma unroll
        for (int p = 0; p < 4; ++p) *(v8bf*)(dst + 8 * p) = rb[rr * 4 + p];
      }
    }
  }
  __syncthreads();                                     // staging dead; reuse LDS

  // ---- write kvs (f32) + load q ----
  sQ[t] = qv[(size_t)b * DD + t];
  {
    int hi = (lane >> 4) << 3, cin = lane & 15;
#pragma unroll
    for (int m = 0; m < 3; ++m)
#pragma unroll
      for (int j = 0; j < 6; ++j) {
        int col = wave * 96 + j * 16 + cin;
#pragma unroll
        for (int v = 0; v < 8; ++v) {
          int row = m * 16 + hi + v;
          if (row < NACT) sKVS[row * K3D + col] = acc[m][j][v];
        }
      }
  }
  __syncthreads();
  // ---- attention scores s[h][n] = (q_h . k_hn)/sqrt(DK) ----
  for (int task = t; task < 16 * NACT; task += 256) {
    int h = task / NACT, n = task % NACT;
    float s = 0.0f;
#pragma unroll
    for (int i = 0; i < 16; ++i) s += sQ[h * 16 + i] * sKVS[n * K3D + h * 16 + i];
    sS[h * NACT + n] = s * 0.25f;
  }
  __syncthreads();
  // ---- per-head softmax (16 heads, 42 keys) ----
  if (t < 16) {
    float m = -3.0e38f;
    for (int n = 0; n < NACT; ++n) m = fmaxf(m, sS[t * NACT + n]);
    float sum = 0.0f;
    for (int n = 0; n < NACT; ++n) { float e = __expf(sS[t * NACT + n] - m); sS[t * NACT + n] = e; sum += e; }
    float inv = 1.0f / sum;
    for (int n = 0; n < NACT; ++n) sS[t * NACT + n] *= inv;
  }
  __syncthreads();
  // ---- mh = attn @ v ----
  {
    int h = t >> 4;
    float a = 0.0f;
    for (int n = 0; n < NACT; ++n) a += sS[h * NACT + n] * sKVS[n * K3D + DD + t];
    sMH[t] = a;
  }
  __syncthreads();
  // ---- mh2 = mh @ W_mhc + b_mhc ----
  {
    float a = 0.0f;
    for (int c = 0; c < DD; ++c) a += sMH[c] * W_mhc[c * DD + t];
    sMH2[t] = a + b_mhc[t];
  }
  __syncthreads();
  // ---- logits, tanh clip ----
  if (t < NACT) {
    float a = 0.0f;
    for (int d = 0; d < DD; ++d) a += sMH2[d] * sKVS[t * K3D + 2 * DD + d];
    sS[t] = 10.0f * tanhf(a * (1.0f / 16.0f));
  }
  __syncthreads();
  if (t == 0) {
    float m = -3.0e38f;
    for (int n = 0; n < NACT; ++n) m = fmaxf(m, sS[n]);
    float sum = 0.0f;
    for (int n = 0; n < NACT; ++n) sum += __expf(sS[n] - m);
    sRed[0] = m; sRed[1] = sum;
  }
  __syncthreads();
  if (t < NACT) out[(size_t)b * NACT + t] = __expf(sS[t] - sRed[0]) / sRed[1];
}

// ---------------- host ----------------
extern "C" void kernel_launch(void* const* d_in, const int* in_sizes, int n_in,
                              void* d_out, int out_size, void* d_ws, size_t ws_size,
                              hipStream_t stream) {
  (void)in_sizes; (void)n_in; (void)out_size; (void)ws_size;
  const float* enc_row   = (const float*)d_in[0];
  const float* enc_col   = (const float*)d_in[1];
  const float* clock_    = (const float*)d_in[2];
  const float* proc_end  = (const float*)d_in[3];
  const float* purge_end = (const float*)d_in[4];
  const float* W_pm_dyn  = (const float*)d_in[5];
  const float* W_pm_cat  = (const float*)d_in[6];
  const float* W_time    = (const float*)d_in[7];
  const float* W_c1      = (const float*)d_in[8];
  const float* b_c1      = (const float*)d_in[9];
  const float* W_c2      = (const float*)d_in[10];
  const float* b_c2      = (const float*)d_in[11];
  const float* Wq1       = (const float*)d_in[12];
  const float* Wk        = (const float*)d_in[13];
  const float* Wv        = (const float*)d_in[14];
  const float* Wshk      = (const float*)d_in[15];
  const float* W_mhc     = (const float*)d_in[16];
  const float* b_mhc     = (const float*)d_in[17];
  const int* pm_lot_idx  = (const int*)d_in[18];
  const int* loc_stage   = (const int*)d_in[19];
  const int* loc_status  = (const int*)d_in[20];
  const int* robot_loc   = (const int*)d_in[21];
  float* out = (float*)d_out;
  char* ws = (char*)d_ws;

  float*  dur     = (float*) (ws + O_DUR);
  __bf16* pm_emb  = (__bf16*)(ws + O_PME);
  __bf16* ctxh    = (__bf16*)(ws + O_CTX);
  __bf16* hid     = (__bf16*)(ws + O_HID);
  __bf16* graphb  = (__bf16*)(ws + O_GR);
  float*  qf      = (float*) (ws + O_Q);
  __bf16* Wpm_t   = (__bf16*)(ws + O_WPM);
  __bf16* Wkvs_t  = (__bf16*)(ws + O_WKVS);
  __bf16* Wc1_t   = (__bf16*)(ws + O_WC1);
  __bf16* Wc2_t   = (__bf16*)(ws + O_WC2);
  __bf16* Wq1_t   = (__bf16*)(ws + O_WQ1);

  auto blocks = [](int n) { return (n + 255) / 256; };

  // K0: weight conversion/transposition to bf16
  wconv_kernel<<<blocks(K3D * DD), 256, 0, stream>>>(W_pm_cat, Wpm_t, K3D, DD, K3D, 0);
  wconv_kernel<<<blocks(K3D * DD), 256, 0, stream>>>(Wk,   Wkvs_t, K3D, DD, K3D, 0);
  wconv_kernel<<<blocks(K3D * DD), 256, 0, stream>>>(Wv,   Wkvs_t, K3D, DD, K3D, 256);
  wconv_kernel<<<blocks(K3D * DD), 256, 0, stream>>>(Wshk, Wkvs_t, K3D, DD, K3D, 512);
  wconv_kernel<<<blocks(K3D * DD), 256, 0, stream>>>(W_c1, Wc1_t, K3D, DD, K3D, 0);
  wconv_kernel<<<blocks(DD * DD),  256, 0, stream>>>(W_c2, Wc2_t, DD, DD, DD, 0);
  wconv_kernel<<<blocks(DD * DD),  256, 0, stream>>>(Wq1,  Wq1_t, DD, DD, DD, 0);

  // K1: scheduling scalars
  dur_kernel<<<blocks(NB), 256, 0, stream>>>(clock_, proc_end, purge_end,
                                             loc_stage, loc_status, robot_loc, dur);

  // K2: pm_emb (M=20480, K=768, N=256)
  pm_emb_gemm<<<dim3((NB * 10) / 128, DD / 64), 256, 0, stream>>>(
      enc_row, enc_col, clock_, proc_end, purge_end, W_pm_dyn,
      pm_lot_idx, loc_stage, Wpm_t, pm_emb);

  // K3: ctx means
  ctx_build<<<NB, 256, 0, stream>>>(enc_row, enc_col, pm_emb, ctxh);

  // K4: ctx -> hid -> graph -> q
  gemm_bf16<<<dim3(NB / 128, DD / 64), 256, 0, stream>>>(
      ctxh, K3D, Wc1_t, K3D, b_c1, 1, nullptr, hid, NB, DD, K3D);
  gemm_bf16<<<dim3(NB / 128, DD / 64), 256, 0, stream>>>(
      hid, DD, Wc2_t, DD, b_c2, 0, nullptr, graphb, NB, DD, DD);
  gemm_bf16<<<dim3(NB / 128, DD / 64), 256, 0, stream>>>(
      graphb, DD, Wq1_t, DD, nullptr, 0, qf, nullptr, NB, DD, DD);

  // K5: fused kvs GEMM + attention + output softmax (one workgroup per batch)
  (void)hipFuncSetAttribute((const void*)fused_kvs_attn,
                            hipFuncAttributeMaxDynamicSharedMemorySize, FUSED_SMEM);
  fused_kvs_attn<<<NB, 256, FUSED_SMEM, stream>>>(
      enc_row, pm_emb, dur, W_time, pm_lot_idx, Wkvs_t, qf, W_mhc, b_mhc, out);
}
